// BaseNetwork_89172111000315
// MI455X (gfx1250) — matrix operand order repaired and checked by reading, not compiled, not run on hardware
//
#include <hip/hip_runtime.h>

// ---------------------------------------------------------------------------
// Types
// ---------------------------------------------------------------------------
typedef _Float16 half_t;
typedef __attribute__((ext_vector_type(16))) _Float16 v16h;
typedef __attribute__((ext_vector_type(8)))  float    v8f;
typedef __attribute__((ext_vector_type(2)))  _Float16 h2_t;
typedef __attribute__((ext_vector_type(2)))  __fp16   fp16x2;
typedef __attribute__((ext_vector_type(4)))  unsigned int u32x4;
typedef __attribute__((ext_vector_type(2)))  float    f32x2;

#define T_DIM 512
#define B_DIM 1024
#define OBS_DIM 128
#define H_DIM 64

// ws layout (halves / bytes)
//  wpack:  half_t[54272]   at byte 0
//    Wb1t [64][128]  @0      Wb2t [64][64] @8192   Wb3t @12288
//    Wit  [192][64]  @16384  Wht  [192][64] @28672
//    Wp1t @40960  Wp2t @45056  Wp3t @49152  Wqt [16][64] @53248
//  xi fragments (u32 words): byte 131072, T*B*192 halves = 50,331,648 words
//  y  fragments (u32 words): byte 131072 + 201326592, T*B*64 halves = 16,777,216 words
#define WS_XI_OFF   (1u << 17)
#define WS_Y_OFF    ((size_t)(1u << 17) + 201326592ull)

// ---------------------------------------------------------------------------
// Helpers
// ---------------------------------------------------------------------------
__device__ __forceinline__ unsigned int pk2(float lo, float hi) {
    fp16x2 h = __builtin_amdgcn_cvt_pkrtz(lo, hi);
    union { fp16x2 h2; unsigned int u; } u; u.h2 = h; return u.u;
}
__device__ __forceinline__ f32x2 unpk2(unsigned int w) {
    union { unsigned int u; h2_t h; } u; u.u = w;
    f32x2 r; r.x = (float)u.h.x; r.y = (float)u.h.y; return r;
}
__device__ __forceinline__ float sigm(float x) { return 1.f / (1.f + __expf(-x)); }
__device__ __forceinline__ float tanh_(float x) {
    float e = __expf(-2.f * x);
    return (1.f - e) / (1.f + e);
}

__device__ __forceinline__ v8f wmma16(v16h a, v16h b, v8f c) {
    // D = A(16x32 f16) * B(32x16 f16) + C(16x16 f32)
    return __builtin_amdgcn_wmma_f32_16x16x32_f16(
        false, a, false, b, (short)0, c, false, false);
}

// A fragment (16x32 f16) from row-major f16 buffer [16][ldk], K-chunk at k0.
__device__ __forceinline__ v16h load_a(const half_t* buf, int ldk, int k0, int lane) {
    int m = lane & 15, khi = lane >> 4;
    const half_t* p = buf + m * ldk + k0 + 8 * khi;
    union { v16h v; unsigned int w[8]; } f;
#pragma unroll
    for (int pp = 0; pp < 8; ++pp) {
        int koff = (pp < 4) ? (2 * pp) : (2 * pp + 8);
        f.w[pp] = *(const unsigned int*)(p + koff);
    }
    return f.v;
}

// A fragment directly from row-major f32 [16][128] (obs tile), cvt to f16
__device__ __forceinline__ v16h load_a_f32(const float* op, int k0, int lane) {
    int m = lane & 15, khi = lane >> 4;
    const float* p = op + m * OBS_DIM + k0 + 8 * khi;
    union { v16h v; unsigned int w[8]; } f;
#pragma unroll
    for (int pp = 0; pp < 8; ++pp) {
        int koff = (pp < 4) ? (2 * pp) : (2 * pp + 8);
        f32x2 v = *(const f32x2*)(p + koff);
        f.w[pp] = pk2(v.x, v.y);
    }
    return f.v;
}

// B fragment (32x16 f16) from column-major weights Wt[ncols][ldk]
__device__ __forceinline__ v16h load_b(const half_t* wt, int ldk, int n0, int k0, int lane) {
    int col = lane & 15, koff = (lane >> 4) * 8;
    const half_t* p = wt + (n0 + col) * ldk + k0 + koff;
    union { v16h v; u32x4 q[2]; } f;
    f.q[0] = *(const u32x4*)(p);
    f.q[1] = *(const u32x4*)(p + 16);
    return f.v;
}

// one Dense(64->64) + relu, activations round-trip through per-wave LDS tile
__device__ __forceinline__ void dense64_relu(half_t* abuf, const half_t* wt,
                                             const float* bias, int lane) {
    int col = lane & 15, khi = lane >> 4;
    v16h a0 = load_a(abuf, 64, 0, lane);
    v16h a1 = load_a(abuf, 64, 32, lane);
    float outv[4][8];
#pragma unroll
    for (int nt = 0; nt < 4; ++nt) {
        v8f c = {0.f,0.f,0.f,0.f,0.f,0.f,0.f,0.f};
        c = wmma16(a0, load_b(wt, 64, nt * 16, 0, lane), c);
        c = wmma16(a1, load_b(wt, 64, nt * 16, 32, lane), c);
        float bv = bias[nt * 16 + col];
#pragma unroll
        for (int j = 0; j < 8; ++j) {
            float x = c[j] + bv;
            outv[nt][j] = fmaxf(x, 0.f);
        }
    }
#pragma unroll
    for (int nt = 0; nt < 4; ++nt)
#pragma unroll
        for (int j = 0; j < 8; ++j)
            abuf[(j + 8 * khi) * 64 + nt * 16 + col] = (half_t)outv[nt][j];
}

// ---------------------------------------------------------------------------
// Kernel 0: pack weight matrix f32 [K][N] -> f16 column-major [N][K]
// ---------------------------------------------------------------------------
__global__ void pack_w_kernel(const float* __restrict__ src, half_t* __restrict__ dst,
                              int K, int N) {
    int i = blockIdx.x * 256 + threadIdx.x;
    if (i >= K * N) return;
    int n = i / K, k = i % K;
    dst[i] = (half_t)src[k * N + n];
}

// ---------------------------------------------------------------------------
// Kernel 1: body MLP + xi = x@Wi + bi ; xi stored as packed f16 C-fragments
// 8 waves/block, 1 tile (16 rows of T*B) per wave. HBM-bound on obs.
// ---------------------------------------------------------------------------
__global__ __launch_bounds__(256) void body_kernel(
    const float* __restrict__ obs,
    const half_t* __restrict__ wpack,
    const float* __restrict__ bb1, const float* __restrict__ bb2,
    const float* __restrict__ bb3, const float* __restrict__ bi,
    unsigned int* __restrict__ xi_words)
{
    __shared__ half_t act[8][1024];   // per-wave 16x64 f16 staging
    int wave = threadIdx.x >> 5, lane = threadIdx.x & 31;
    int col = lane & 15, khi = lane >> 4;
    long tile = (long)blockIdx.x * 8 + wave;          // 0..32767
    half_t* abuf = &act[wave][0];

    const half_t* Wb1t = wpack;            // [64][128]
    const half_t* Wb2t = wpack + 8192;     // [64][64]
    const half_t* Wb3t = wpack + 12288;    // [64][64]
    const half_t* Wit  = wpack + 16384;    // [192][64]

    const float* op = obs + tile * 16 * OBS_DIM;

    // ---- layer 1: OBS(128) -> 64, A frags straight from global f32 ----
    v16h a4[4];
#pragma unroll
    for (int kc = 0; kc < 4; ++kc) a4[kc] = load_a_f32(op, kc * 32, lane);

    float outv[4][8];
#pragma unroll
    for (int nt = 0; nt < 4; ++nt) {
        v8f c = {0.f,0.f,0.f,0.f,0.f,0.f,0.f,0.f};
#pragma unroll
        for (int kc = 0; kc < 4; ++kc)
            c = wmma16(a4[kc], load_b(Wb1t, 128, nt * 16, kc * 32, lane), c);
        float bv = bb1[nt * 16 + col];
#pragma unroll
        for (int j = 0; j < 8; ++j) outv[nt][j] = fmaxf(c[j] + bv, 0.f);
    }
#pragma unroll
    for (int nt = 0; nt < 4; ++nt)
#pragma unroll
        for (int j = 0; j < 8; ++j)
            abuf[(j + 8 * khi) * 64 + nt * 16 + col] = (half_t)outv[nt][j];

    // ---- layers 2 & 3 ----
    dense64_relu(abuf, Wb2t, bb2, lane);
    dense64_relu(abuf, Wb3t, bb3, lane);

    // ---- xi = x @ Wi + bi  (N = 192), store packed fragments ----
    v16h a0 = load_a(abuf, 64, 0, lane);
    v16h a1 = load_a(abuf, 64, 32, lane);
    unsigned int* xw = xi_words + tile * (12 * 128);
#pragma unroll
    for (int nt = 0; nt < 12; ++nt) {
        v8f c = {0.f,0.f,0.f,0.f,0.f,0.f,0.f,0.f};
        c = wmma16(a0, load_b(Wit, 64, nt * 16, 0, lane), c);
        c = wmma16(a1, load_b(Wit, 64, nt * 16, 32, lane), c);
        float bv = bi[nt * 16 + col];
#pragma unroll
        for (int j2 = 0; j2 < 4; ++j2)
            xw[nt * 128 + j2 * 32 + lane] = pk2(c[2 * j2] + bv, c[2 * j2 + 1] + bv);
    }
}

// ---------------------------------------------------------------------------
// Kernel 2: GRU scan. One wave owns 16 batch rows for all 512 steps.
// h in registers (C layout, f32); per step LDS transpose -> A frag.
// Wh staged in LDS once: inner-loop B-fragments are ds_load_b128 pairs,
// keeping the serial dependence chain off the vmem latency path.
// ---------------------------------------------------------------------------
__global__ __launch_bounds__(128) void gru_kernel(
    const float* __restrict__ hidden,
    const unsigned char* __restrict__ dones,
    const half_t* __restrict__ Wht,          // [192][64] f16 col-major (global)
    const float* __restrict__ bhn,
    const unsigned int* __restrict__ xi_words,
    unsigned int* __restrict__ y_words,
    float* __restrict__ h_out)
{
    __shared__ half_t wlds[12288];           // Wh [192][64] f16, 24 KB
    __shared__ half_t hstage[4][1024];       // per-wave 16x64 h staging
    int wave = threadIdx.x >> 5, lane = threadIdx.x & 31;
    int col = lane & 15, khi = lane >> 4;
    int btile = blockIdx.x * 4 + wave;        // 0..63
    int b0 = btile * 16;
    half_t* hb = &hstage[wave][0];

    // cooperative stage of Wh into LDS (1536 x b128)
    {
        const u32x4* src = (const u32x4*)Wht;
        u32x4* dst = (u32x4*)wlds;
#pragma unroll
        for (int i = 0; i < 12; ++i)
            dst[threadIdx.x + i * 128] = src[threadIdx.x + i * 128];
    }
    __syncthreads();

    float hc[4][8];
#pragma unroll
    for (int g = 0; g < 4; ++g)
#pragma unroll
        for (int j = 0; j < 8; ++j)
            hc[g][j] = hidden[(b0 + j + 8 * khi) * 64 + g * 16 + col];

    float bn[4];
#pragma unroll
    for (int g = 0; g < 4; ++g) bn[g] = bhn[g * 16 + col];

    for (int t = 0; t < T_DIM; ++t) {
        // reset carry on done
        const unsigned char* dp = dones + (long)t * B_DIM + b0 + 8 * khi;
#pragma unroll
        for (int j = 0; j < 8; ++j) {
            if (dp[j]) {
                hc[0][j] = 0.f; hc[1][j] = 0.f; hc[2][j] = 0.f; hc[3][j] = 0.f;
            }
        }
        // stage h (f16) for A-fragment pickup (same-wave DS ops are in order)
#pragma unroll
        for (int g = 0; g < 4; ++g)
#pragma unroll
            for (int j = 0; j < 8; ++j)
                hb[(j + 8 * khi) * 64 + g * 16 + col] = (half_t)hc[g][j];

        v16h a0 = load_a(hb, 64, 0, lane);
        v16h a1 = load_a(hb, 64, 32, lane);

        const unsigned int* xw = xi_words + ((long)t * 64 + btile) * 1536;
        unsigned int* yw = y_words + ((long)t * 64 + btile) * 512;
        // prefetch next step's xi fragments
        __builtin_prefetch(xw + 64 * 1536, 0, 1);

        float rg[4][8];
#pragma unroll
        for (int g = 0; g < 4; ++g) {   // r gate: cols 0..63
            v8f c = {0.f,0.f,0.f,0.f,0.f,0.f,0.f,0.f};
            c = wmma16(a0, load_b(wlds, 64, g * 16, 0, lane), c);
            c = wmma16(a1, load_b(wlds, 64, g * 16, 32, lane), c);
#pragma unroll
            for (int j2 = 0; j2 < 4; ++j2) {
                f32x2 x = unpk2(xw[g * 128 + j2 * 32 + lane]);
                rg[g][2 * j2]     = sigm(x.x + c[2 * j2]);
                rg[g][2 * j2 + 1] = sigm(x.y + c[2 * j2 + 1]);
            }
        }
#pragma unroll
        for (int g = 0; g < 4; ++g) {   // z: cols 64..127, n: cols 128..191
            v8f cz = {0.f,0.f,0.f,0.f,0.f,0.f,0.f,0.f};
            v8f cn = {0.f,0.f,0.f,0.f,0.f,0.f,0.f,0.f};
            cz = wmma16(a0, load_b(wlds, 64, (g + 4) * 16, 0, lane), cz);
            cz = wmma16(a1, load_b(wlds, 64, (g + 4) * 16, 32, lane), cz);
            cn = wmma16(a0, load_b(wlds, 64, (g + 8) * 16, 0, lane), cn);
            cn = wmma16(a1, load_b(wlds, 64, (g + 8) * 16, 32, lane), cn);
#pragma unroll
            for (int j2 = 0; j2 < 4; ++j2) {
                f32x2 xz = unpk2(xw[(g + 4) * 128 + j2 * 32 + lane]);
                f32x2 xn = unpk2(xw[(g + 8) * 128 + j2 * 32 + lane]);
                float z0 = sigm(xz.x + cz[2 * j2]);
                float z1 = sigm(xz.y + cz[2 * j2 + 1]);
                float n0 = tanh_(xn.x + rg[g][2 * j2]     * (cn[2 * j2]     + bn[g]));
                float n1 = tanh_(xn.y + rg[g][2 * j2 + 1] * (cn[2 * j2 + 1] + bn[g]));
                float h0 = (1.f - z0) * n0 + z0 * hc[g][2 * j2];
                float h1 = (1.f - z1) * n1 + z1 * hc[g][2 * j2 + 1];
                hc[g][2 * j2] = h0;
                hc[g][2 * j2 + 1] = h1;
                yw[g * 128 + j2 * 32 + lane] = pk2(h0, h1);
            }
        }
    }
    // h_last -> d_out[0 .. B*H)
#pragma unroll
    for (int g = 0; g < 4; ++g)
#pragma unroll
        for (int j = 0; j < 8; ++j)
            h_out[(b0 + j + 8 * khi) * 64 + g * 16 + col] = hc[g][j];
}

// ---------------------------------------------------------------------------
// Kernel 3: post MLP (3x Dense64+relu) + Q head (N=16), q f32 to d_out
// ---------------------------------------------------------------------------
__global__ __launch_bounds__(256) void post_kernel(
    const unsigned int* __restrict__ y_words,
    const half_t* __restrict__ wpost,   // Wp1t @0, Wp2t @4096, Wp3t @8192, Wqt @12288
    const float* __restrict__ bp1, const float* __restrict__ bp2,
    const float* __restrict__ bp3, const float* __restrict__ bq,
    float* __restrict__ q_out)
{
    __shared__ half_t act[8][1024];
    int wave = threadIdx.x >> 5, lane = threadIdx.x & 31;
    int col = lane & 15, khi = lane >> 4;
    long tile = (long)blockIdx.x * 8 + wave;
    half_t* abuf = &act[wave][0];

    // unpack y fragments into row-major f16 [16][64]
    const unsigned int* yw = y_words + tile * 512;
#pragma unroll
    for (int nt = 0; nt < 4; ++nt)
#pragma unroll
        for (int j2 = 0; j2 < 4; ++j2) {
            f32x2 v = unpk2(yw[nt * 128 + j2 * 32 + lane]);
            abuf[(2 * j2     + 8 * khi) * 64 + nt * 16 + col] = (half_t)v.x;
            abuf[(2 * j2 + 1 + 8 * khi) * 64 + nt * 16 + col] = (half_t)v.y;
        }

    dense64_relu(abuf, wpost,        bp1, lane);
    dense64_relu(abuf, wpost + 4096, bp2, lane);
    dense64_relu(abuf, wpost + 8192, bp3, lane);

    // Q head: 64 -> 16
    v16h a0 = load_a(abuf, 64, 0, lane);
    v16h a1 = load_a(abuf, 64, 32, lane);
    v8f c = {0.f,0.f,0.f,0.f,0.f,0.f,0.f,0.f};
    c = wmma16(a0, load_b(wpost + 12288, 64, 0, 0, lane), c);
    c = wmma16(a1, load_b(wpost + 12288, 64, 0, 32, lane), c);
    float bv = bq[col];
    long rowb = tile * 16;
#pragma unroll
    for (int j = 0; j < 8; ++j)
        q_out[(rowb + j + 8 * khi) * 16 + col] = c[j] + bv;
}

// ---------------------------------------------------------------------------
// Launch
// ---------------------------------------------------------------------------
extern "C" void kernel_launch(void* const* d_in, const int* in_sizes, int n_in,
                              void* d_out, int out_size, void* d_ws, size_t ws_size,
                              hipStream_t stream) {
    const float* hidden = (const float*)d_in[0];
    const float* obs    = (const float*)d_in[1];
    const unsigned char* dones = (const unsigned char*)d_in[2];
    const float* Wb1 = (const float*)d_in[3];  const float* bb1 = (const float*)d_in[4];
    const float* Wb2 = (const float*)d_in[5];  const float* bb2 = (const float*)d_in[6];
    const float* Wb3 = (const float*)d_in[7];  const float* bb3 = (const float*)d_in[8];
    const float* Wi  = (const float*)d_in[9];  const float* bi  = (const float*)d_in[10];
    const float* Wh  = (const float*)d_in[11]; const float* bhn = (const float*)d_in[12];
    const float* Wp1 = (const float*)d_in[13]; const float* bp1 = (const float*)d_in[14];
    const float* Wp2 = (const float*)d_in[15]; const float* bp2 = (const float*)d_in[16];
    const float* Wp3 = (const float*)d_in[17]; const float* bp3 = (const float*)d_in[18];
    const float* Wq  = (const float*)d_in[19]; const float* bq  = (const float*)d_in[20];

    half_t* wpack = (half_t*)d_ws;
    unsigned int* xi_words = (unsigned int*)((char*)d_ws + WS_XI_OFF);
    unsigned int* y_words  = (unsigned int*)((char*)d_ws + WS_Y_OFF);
    float* out = (float*)d_out;

    // pack all weights -> f16 column-major
    pack_w_kernel<<<(128 * 64 + 255) / 256, 256, 0, stream>>>(Wb1, wpack,          128, 64);
    pack_w_kernel<<<( 64 * 64 + 255) / 256, 256, 0, stream>>>(Wb2, wpack + 8192,    64, 64);
    pack_w_kernel<<<( 64 * 64 + 255) / 256, 256, 0, stream>>>(Wb3, wpack + 12288,   64, 64);
    pack_w_kernel<<<(64 * 192 + 255) / 256, 256, 0, stream>>>(Wi,  wpack + 16384,   64, 192);
    pack_w_kernel<<<(64 * 192 + 255) / 256, 256, 0, stream>>>(Wh,  wpack + 28672,   64, 192);
    pack_w_kernel<<<( 64 * 64 + 255) / 256, 256, 0, stream>>>(Wp1, wpack + 40960,   64, 64);
    pack_w_kernel<<<( 64 * 64 + 255) / 256, 256, 0, stream>>>(Wp2, wpack + 45056,   64, 64);
    pack_w_kernel<<<( 64 * 64 + 255) / 256, 256, 0, stream>>>(Wp3, wpack + 49152,   64, 64);
    pack_w_kernel<<<( 64 * 16 + 255) / 256, 256, 0, stream>>>(Wq,  wpack + 53248,   64, 16);

    // body MLP + xi projection: 32768 row-tiles, 8 per block
    body_kernel<<<4096, 256, 0, stream>>>(obs, wpack, bb1, bb2, bb3, bi, xi_words);

    // sequential GRU scan: 64 batch-tiles, 4 waves per block
    gru_kernel<<<16, 128, 0, stream>>>(hidden, dones, wpack + 28672, bhn,
                                       xi_words, y_words, out);

    // post MLP + Q head
    post_kernel<<<4096, 256, 0, stream>>>(y_words, wpack + 40960,
                                          bp1, bp2, bp3, bq, out + B_DIM * H_DIM);
}
